// SpatialBertSelfAttention_78331613544904
// MI455X (gfx1250) — matrix operand order repaired and checked by reading, not compiled, not run on hardware
//
#include <hip/hip_runtime.h>
#include <hip/hip_bf16.h>

// ---------------------------------------------------------------------------
// SpatialBertSelfAttention for MI455X (gfx1250, wave32, WMMA f16 16x16x32)
//   BS=128, N=182 (pad->192), D=768, H=12, d=64, S=20 text, R=150 ocr/obj
// ---------------------------------------------------------------------------

typedef __attribute__((ext_vector_type(16))) _Float16     v16h;
typedef __attribute__((ext_vector_type(8)))  float        v8f;
typedef __attribute__((ext_vector_type(8)))  unsigned int v8u;

union FragH { v8u u; v16h h; };
union Pack8H { uint4 q; _Float16 hh[8]; };

#define SEQ    182
#define NPAD   192
#define DMODEL 768
#define NROWS  23296   // 128 * 182
#define HEADS  12
#define HDIM   64
#define NEGM   (-10000.0f)

static __device__ __forceinline__ v8f wmma_f16(v16h a, v16h b, v8f c) {
  // D = A(16x32 f16) * B(32x16 f16) + C(16x16 f32)
  return __builtin_amdgcn_wmma_f32_16x16x32_f16(
      /*neg_a=*/false, a, /*neg_b=*/false, b,
      /*c_mod=*/(short)0, c, /*reuse_a=*/false, /*reuse_b=*/false);
}

// --------------------------- f32 -> f16 convert ----------------------------
__global__ __launch_bounds__(256) void cvt_f32_to_f16(
    const float* __restrict__ in, _Float16* __restrict__ out, int n) {
  int i = blockIdx.x * blockDim.x + threadIdx.x;
  if (i < n) out[i] = (_Float16)in[i];
}

// ------------------------------ QKV GEMM -----------------------------------
// Y[g] = X (23296x768) * W[g]^T + b[g], stored f16.
// Wave tile: 32x64 (2 M-tiles x 4 N-tiles) -> 8 WMMAs per 32-K step, B reuse x2.
// grid: x = M-tile-pair group (8 waves/block), y = 64-wide N strip, z = {q,k,v}
__global__ __launch_bounds__(256) void qkv_gemm_kernel(
    const _Float16* __restrict__ X, const _Float16* __restrict__ W,
    const float* __restrict__ bq, const float* __restrict__ bk,
    const float* __restrict__ bv, _Float16* __restrict__ Y) {
  const int lane   = threadIdx.x & 31;
  const int wave   = threadIdx.x >> 5;
  const int laneHi = lane >> 4;
  const int l15    = lane & 15;
  const int mt0    = (blockIdx.x * 8 + wave) * 2;  // 0..1454 step 2
  const int nbase  = blockIdx.y * 64;              // 0..767 step 64
  const int g      = blockIdx.z;                   // 0=q 1=k 2=v

  const _Float16* Wg = W + (size_t)g * DMODEL * DMODEL;
  _Float16*       Yg = Y + (size_t)g * NROWS * DMODEL;
  const float*    bg = (g == 0) ? bq : ((g == 1) ? bk : bv);

  // ISA f16 A(16x32) / B(32x16) per-lane half offsets (pairs -> b32 loads)
  int aoff[8], boff[8];
#pragma unroll
  for (int j = 0; j < 8; ++j) {
    aoff[j] = ((j < 4) ? 2 * j : 16 + 2 * (j - 4)) + laneHi * 8;
    boff[j] = ((j < 4) ? 2 * j : 16 + 2 * (j - 4)) + laneHi * 8;
  }

  const unsigned int* arow0 =
      (const unsigned int*)(X + ((size_t)mt0 * 16 + l15) * DMODEL);
  const unsigned int* arow1 =
      (const unsigned int*)(X + ((size_t)(mt0 + 1) * 16 + l15) * DMODEL);
  const unsigned int* brow[4];
#pragma unroll
  for (int nt = 0; nt < 4; ++nt)
    brow[nt] =
        (const unsigned int*)(Wg + (size_t)(nbase + nt * 16 + l15) * DMODEL);

  v8f acc0[4] = {};
  v8f acc1[4] = {};
  for (int ks = 0; ks < DMODEL; ks += 32) {
    FragH a0, a1;
#pragma unroll
    for (int j = 0; j < 8; ++j) {
      a0.u[j] = arow0[(ks + aoff[j]) >> 1];
      a1.u[j] = arow1[(ks + aoff[j]) >> 1];
    }
#pragma unroll
    for (int nt = 0; nt < 4; ++nt) {
      FragH bf;
#pragma unroll
      for (int j = 0; j < 8; ++j) bf.u[j] = brow[nt][(ks + boff[j]) >> 1];
      acc0[nt] = wmma_f16(a0.h, bf.h, acc0[nt]);
      acc1[nt] = wmma_f16(a1.h, bf.h, acc1[nt]);
    }
  }

#pragma unroll
  for (int nt = 0; nt < 4; ++nt) {
    const int   col = nbase + nt * 16 + l15;
    const float bvv = bg[col];
#pragma unroll
    for (int j = 0; j < 8; ++j) {
      const int row0 = mt0 * 16 + j + laneHi * 8;          // C-layout row
      const int row1 = (mt0 + 1) * 16 + j + laneHi * 8;
      Yg[(size_t)row0 * DMODEL + col] = (_Float16)(acc0[nt][j] + bvv);
      Yg[(size_t)row1 * DMODEL + col] = (_Float16)(acc1[nt][j] + bvv);
    }
  }
}

// ------------------------------ Attention ----------------------------------
// One block per (b,h): 384 threads = 12 waves; wave w owns query tile w.
// LDS: V^T (64 x 192 f16, 24KB) + per-wave P (16 x 192 f16, 12x6KB) = 96KB.
__global__ __launch_bounds__(384) void spatial_attn_kernel(
    const _Float16* __restrict__ QKV, const float* __restrict__ am,
    const float* __restrict__ adj, float* __restrict__ out) {
  extern __shared__ char smem[];
  _Float16* vt   = (_Float16*)smem;         // [64][192] V transposed
  _Float16* pbuf = vt + 64 * NPAD;          // 12 waves x [16][192]

  const int b      = blockIdx.x;
  const int h      = blockIdx.y;
  const int tid    = threadIdx.x;
  const int wave   = tid >> 5;              // query tile 0..11
  const int lane   = tid & 31;
  const int laneHi = lane >> 4;
  const int l15    = lane & 15;

  const _Float16* Q = QKV;
  const _Float16* K = QKV + (size_t)NROWS * DMODEL;
  const _Float16* V = QKV + (size_t)2 * NROWS * DMODEL;
  const size_t rowbase = (size_t)b * SEQ * DMODEL + (size_t)h * HDIM;

  // ---- stage V^T into LDS: 8 consecutive threads read one contiguous
  //      128-byte chunk of a V row (coalesced b128), scatter-transpose via LDS.
  for (int idx = tid; idx < NPAD * 8; idx += 384) {
    const int key = idx >> 3;               // 0..191
    const int d0  = (idx & 7) * 8;          // 0..56 step 8
    Pack8H u;
    if (key < SEQ)
      u.q = *(const uint4*)(V + rowbase + (size_t)key * DMODEL + d0);
    else
      u.q = make_uint4(0u, 0u, 0u, 0u);
#pragma unroll
    for (int i = 0; i < 8; ++i) vt[(d0 + i) * NPAD + key] = u.hh[i];
  }
  __syncthreads();

  int aoff[8], boff[8];
#pragma unroll
  for (int j = 0; j < 8; ++j) {
    aoff[j] = ((j < 4) ? 2 * j : 16 + 2 * (j - 4)) + laneHi * 8;
    boff[j] = ((j < 4) ? 2 * j : 16 + 2 * (j - 4)) + laneHi * 8;
  }

  // ---- scores: S = Q * K^T  (12 k-tiles, d split 2x32) ----
  v8f acc[12] = {};
  {
    int qr = wave * 16 + l15;               // A-layout row
    if (qr >= SEQ) qr = SEQ - 1;
    const unsigned int* qrow_u =
        (const unsigned int*)(Q + rowbase + (size_t)qr * DMODEL);
#pragma unroll
    for (int ds = 0; ds < HDIM; ds += 32) {
      FragH a;
#pragma unroll
      for (int j = 0; j < 8; ++j) a.u[j] = qrow_u[(ds + aoff[j]) >> 1];
#pragma unroll
      for (int t = 0; t < 12; ++t) {
        int kn = t * 16 + l15;              // B-layout column = key row
        if (kn >= SEQ) kn = SEQ - 1;
        const unsigned int* krow_u =
            (const unsigned int*)(K + rowbase + (size_t)kn * DMODEL);
        FragH bf;
#pragma unroll
        for (int j = 0; j < 8; ++j) bf.u[j] = krow_u[(ds + boff[j]) >> 1];
        acc[t] = wmma_f16(a.h, bf.h, acc[t]);
      }
    }
  }

  // ---- scale + combined mask (quadrant rule + adjacency + padding mask) ----
  float cmbmax[8];
#pragma unroll
  for (int j = 0; j < 8; ++j) cmbmax[j] = -3.0e4f;
#pragma unroll
  for (int t = 0; t < 12; ++t) {
    const int   kc  = t * 16 + l15;                 // key column
    const float amv = (kc < SEQ) ? am[(size_t)b * SEQ + kc] : NEGM;
#pragma unroll
    for (int j = 0; j < 8; ++j) {
      const int qr2 = wave * 16 + j + laneHi * 8;   // query row (C layout)
      float sp;
      if (qr2 >= 170) {                              // decode query (or pad)
        sp = NEGM;
      } else if (qr2 < 20) {                         // text query
        sp = (kc >= 170 && kc < SEQ) ? 0.0f : NEGM;
      } else {                                       // ocr/obj query
        if (kc < 20 || kc >= SEQ) sp = NEGM;
        else if (kc < 170)
          sp = (adj[(((size_t)b * 150 + (qr2 - 20)) * 150 + (kc - 20)) * HEADS +
                    h] != 0.0f)
                   ? 0.0f
                   : NEGM;
        else sp = 0.0f;                              // decode keys attendable
      }
      const float cmb = fminf(amv, sp);
      acc[t][j]  = acc[t][j] * 0.125f + cmb;         // /sqrt(64) + mask
      cmbmax[j]  = fmaxf(cmbmax[j], cmb);
    }
  }

  // ---- row softmax * entity mask (reduce across 12 tiles + 16 lanes) ----
#pragma unroll
  for (int j = 0; j < 8; ++j) {
    float m = -3.4e38f;
#pragma unroll
    for (int t = 0; t < 12; ++t) m = fmaxf(m, acc[t][j]);
    float cm = cmbmax[j];
#pragma unroll
    for (int mk = 1; mk <= 8; mk <<= 1) {
      m  = fmaxf(m, __shfl_xor(m, mk));
      cm = fmaxf(cm, __shfl_xor(cm, mk));
    }
    float sum = 0.0f;
#pragma unroll
    for (int t = 0; t < 12; ++t) {
      const float e = __expf(acc[t][j] - m);
      acc[t][j] = e;
      sum += e;
    }
#pragma unroll
    for (int mk = 1; mk <= 8; mk <<= 1) sum += __shfl_xor(sum, mk);
    const float scale = ((cm > -5000.0f) ? 1.0f : 0.0f) / sum;  // entity mask
#pragma unroll
    for (int t = 0; t < 12; ++t) acc[t][j] *= scale;
  }

  // ---- write P (f16) to this wave's LDS region, row-major 16x192 ----
  _Float16* pb = pbuf + (size_t)wave * 16 * NPAD;
#pragma unroll
  for (int t = 0; t < 12; ++t)
#pragma unroll
    for (int j = 0; j < 8; ++j)
      pb[(j + laneHi * 8) * NPAD + t * 16 + l15] = (_Float16)acc[t][j];

  // ---- ctx = P (16x192) * V (192x64): 6 k-steps x 4 d-tiles ----
  v8f dacc[4] = {};
  const unsigned int* pb_u = (const unsigned int*)pb;
  const unsigned int* vt_u = (const unsigned int*)vt;
#pragma unroll
  for (int kc = 0; kc < NPAD; kc += 32) {
    FragH a;
#pragma unroll
    for (int j = 0; j < 8; ++j)
      a.u[j] = pb_u[(l15 * NPAD + kc + aoff[j]) >> 1];
#pragma unroll
    for (int dt = 0; dt < 4; ++dt) {
      const int d = dt * 16 + l15;          // B column = output feature
      FragH bf;
#pragma unroll
      for (int j = 0; j < 8; ++j)
        bf.u[j] = vt_u[(d * NPAD + kc + boff[j]) >> 1];
      dacc[dt] = wmma_f16(a.h, bf.h, dacc[dt]);
    }
  }

  // ---- store ctx f32: (b, qrow, h*64 + d) ----
#pragma unroll
  for (int dt = 0; dt < 4; ++dt)
#pragma unroll
    for (int j = 0; j < 8; ++j) {
      const int qrow = wave * 16 + j + laneHi * 8;
      if (qrow < SEQ)
        out[((size_t)b * SEQ + qrow) * DMODEL + h * HDIM + dt * 16 + l15] =
            dacc[dt][j];
    }
}

// ------------------------------ launcher -----------------------------------
extern "C" void kernel_launch(void* const* d_in, const int* in_sizes, int n_in,
                              void* d_out, int out_size, void* d_ws,
                              size_t ws_size, hipStream_t stream) {
  const float* hidden = (const float*)d_in[0];
  const float* amask  = (const float*)d_in[1];
  const float* adj    = (const float*)d_in[2];
  const float* Wq     = (const float*)d_in[3];
  const float* bq     = (const float*)d_in[4];
  const float* Wk     = (const float*)d_in[5];
  const float* bk     = (const float*)d_in[6];
  const float* Wv     = (const float*)d_in[7];
  const float* bv     = (const float*)d_in[8];
  float*       out    = (float*)d_out;

  // workspace layout (f16): X16 | W16[3] | QKV16[3]
  _Float16* X16   = (_Float16*)d_ws;
  _Float16* W16   = X16 + (size_t)NROWS * DMODEL;
  _Float16* QKV16 = W16 + (size_t)3 * DMODEL * DMODEL;

  const int nX = NROWS * DMODEL;      // 17,891,328
  const int nW = DMODEL * DMODEL;     // 589,824
  cvt_f32_to_f16<<<(nX + 255) / 256, 256, 0, stream>>>(hidden, X16, nX);
  cvt_f32_to_f16<<<(nW + 255) / 256, 256, 0, stream>>>(Wq, W16, nW);
  cvt_f32_to_f16<<<(nW + 255) / 256, 256, 0, stream>>>(Wk, W16 + nW, nW);
  cvt_f32_to_f16<<<(nW + 255) / 256, 256, 0, stream>>>(Wv, W16 + 2 * (size_t)nW, nW);

  dim3 g1(91, 12, 3);   // 1456 M-tiles / 2 per wave / 8 waves, 12 strips, qkv
  qkv_gemm_kernel<<<g1, 256, 0, stream>>>(X16, W16, bq, bk, bv, QKV16);

  dim3 g2(128, HEADS);  // (batch, head)
  const size_t lds = (size_t)(HDIM * NPAD + HEADS * 16 * NPAD) * sizeof(_Float16);
  spatial_attn_kernel<<<g2, 384, lds, stream>>>(QKV16, amask, adj, out);
}
